// PriorKnowledgeLayer_68418829025670
// MI455X (gfx1250) — matrix operand order repaired and checked by reading, not compiled, not run on hardware
//
#include <hip/hip_runtime.h>
#include <hip/hip_bf16.h>

// Problem dims (fixed by the reference)
#define B_DIM 64
#define S_DIM 2048
#define V_DIM 256
#define TILE_S 64                  // output rows per tile
#define TILES_PER_WG 2             // two tiles per WG -> prefetch overlap
#define MT_STRIDE 264              // 256 + 8 bf16 pad -> kills LDS bank conflicts
#define DA_ROWS (TILE_S + 1)       // need rows base..base+64
#define MT_ELEMS (V_DIM * MT_STRIDE)
#define DA_ELEMS (DA_ROWS * MT_STRIDE)
#define MT_BYTES (MT_ELEMS * 2)    // 135168 B = 132 KB
#define SMEM_BYTES (MT_BYTES + DA_ELEMS * 2 + 2 * TILE_S * 4)

typedef __attribute__((ext_vector_type(16))) __bf16 v16bf;
typedef __attribute__((ext_vector_type(8)))  __bf16 v8bf;
typedef __attribute__((ext_vector_type(8)))  float  v8f;

union ABFrag { v16bf v; v8bf h[2]; };

// Scheduler fence the AMDGPU machine scheduler must honor (mask 0 = nothing
// may be reordered across).  Empty inline asm proved insufficient in round 4.
#if __has_builtin(__builtin_amdgcn_sched_barrier)
#define SCHED_FENCE() __builtin_amdgcn_sched_barrier(0)
#else
#define SCHED_FENCE() asm volatile("" ::: "memory")
#endif

// ---------------------------------------------------------------------------
// Prep kernel (one-time, tiny): M (f32, row-major) -> d_ws as bf16, transposed
// and padded to MT_STRIDE, i.e. exactly the LDS image the main kernel wants.
// ---------------------------------------------------------------------------
__global__ __launch_bounds__(256)
void prep_mt_kernel(const float* __restrict__ M, __bf16* __restrict__ MtG)
{
    const int n = threadIdx.x;                 // column of M == row of Mt
#pragma unroll
    for (int kk = 0; kk < 4; ++kk) {
        const int k = blockIdx.x * 4 + kk;     // row of M (K index)
        const float v = M[k * V_DIM + n];      // coalesced read across threads
        MtG[n * MT_STRIDE + k] = (__bf16)v;    // transposed write (tiny total)
    }
    if (blockIdx.x == 0) {                     // defined values in the pad lanes
#pragma unroll
        for (int p = 0; p < 8; ++p)
            MtG[n * MT_STRIDE + V_DIM + p] = (__bf16)0.f;
    }
}

// ---------------------------------------------------------------------------
// Main fused kernel.  Per WG: stage Mt (async-to-LDS), then for each of two
// S-tiles: log-softmax -> LDS, bf16 WMMA GEMM (d @ M), per-row dot with
// d[i+1].  While tile 0's GEMM runs, tile 1's logits are pulled into cache
// with global_prefetch_b8 (no counters, pure memory-system overlap).
// ---------------------------------------------------------------------------
template <bool USE_WS>
__global__ __launch_bounds__(256, 1)
void pk_bilinear_wmma_kernel(const float* __restrict__ logits,
                             const float* __restrict__ M,
                             const void*  __restrict__ MtG,
                             float* __restrict__ out)
{
    extern __shared__ char smem[];
    __bf16* Mt   = (__bf16*)smem;                 // [V][MT_STRIDE], M^T (bf16)
    __bf16* dA   = Mt + MT_ELEMS;                 // [DA_ROWS][MT_STRIDE], log-softmax rows
    float*  part = (float*)(dA + DA_ELEMS);       // [2][TILE_S] partial dot results

    const int tid   = threadIdx.x;
    const int lane  = tid & 31;
    const int wave  = tid >> 5;
    const int b     = blockIdx.y;
    const int base0 = blockIdx.x * (TILE_S * TILES_PER_WG);

    const int rt      = wave & 3;      // 16-row tile
    const int nh      = wave >> 2;     // 128-column half
    const int half    = lane >> 4;
    const int l16     = lane & 15;
    const int rowbase = 16 * rt;
    const int arow    = rowbase + l16;

    for (int sub = 0; sub < TILES_PER_WG; ++sub) {
        const int base = base0 + sub * TILE_S;

        // ---- Phase 0 (sub==0 only): stage M^T (bf16) into LDS ----
        if (sub == 0) {
            if (USE_WS) {
                // 132 KB byte-copy via global_load_async_to_lds_b128 (ASYNCcnt),
                // overlapped with the log-softmax phase below.
                const uint64_t gbase = (uint64_t)(uintptr_t)MtG;
                const uint32_t lbase = (uint32_t)(uintptr_t)(void*)Mt;
                for (int it = 0; it < (MT_BYTES / 16) / 256; ++it) {
                    const uint32_t off = (uint32_t)(it * 256 + tid) * 16u;
                    const uint64_t g   = gbase + off;
                    const uint32_t l   = lbase + off;
                    asm volatile("global_load_async_to_lds_b128 %0, %1, off"
                                 :: "v"(l), "v"(g) : "memory");
                }
            } else {
                const float4* M4 = (const float4*)M;
                for (int idx = tid; idx < V_DIM * (V_DIM / 4); idx += 256) {
                    const int k  = idx >> 6;
                    const int g  = idx & 63;
                    const float4 f = M4[k * 64 + g];
                    const int n0 = g * 4;
                    Mt[(n0 + 0) * MT_STRIDE + k] = (__bf16)f.x;
                    Mt[(n0 + 1) * MT_STRIDE + k] = (__bf16)f.y;
                    Mt[(n0 + 2) * MT_STRIDE + k] = (__bf16)f.z;
                    Mt[(n0 + 3) * MT_STRIDE + k] = (__bf16)f.w;
                }
            }
        }

        // ---- Phase 1: log-softmax of rows [base, base+64] -> bf16 in LDS ----
        for (int r = wave; r < DA_ROWS; r += 8) {
            const int srow = base + r;
            if (srow < S_DIM) {
                const float* rowp = logits + ((size_t)b * S_DIM + srow) * V_DIM;
                float x[8];
#pragma unroll
                for (int k = 0; k < 8; ++k) x[k] = rowp[lane + 32 * k];
                float m = x[0];
#pragma unroll
                for (int k = 1; k < 8; ++k) m = fmaxf(m, x[k]);
#pragma unroll
                for (int w = 16; w >= 1; w >>= 1) m = fmaxf(m, __shfl_xor(m, w, 32));
                float s = 0.f;
#pragma unroll
                for (int k = 0; k < 8; ++k) s += __expf(x[k] - m);
#pragma unroll
                for (int w = 16; w >= 1; w >>= 1) s += __shfl_xor(s, w, 32);
                const float logZ = m + __logf(s);
#pragma unroll
                for (int k = 0; k < 8; ++k)
                    dA[r * MT_STRIDE + lane + 32 * k] = (__bf16)(x[k] - logZ);
            } else {
#pragma unroll
                for (int k = 0; k < 8; ++k)
                    dA[r * MT_STRIDE + lane + 32 * k] = (__bf16)0.f;
            }
        }

        if (USE_WS && sub == 0) {
            // Async-to-LDS completion is NOT covered by the barrier.
#if __has_builtin(__builtin_amdgcn_s_wait_asynccnt)
            __builtin_amdgcn_s_wait_asynccnt(0);
#else
            asm volatile("s_wait_asynccnt 0x0" ::: "memory");
#endif
        }
        __syncthreads();

        // ---- Prefetch next tile's logits while this tile's GEMM runs ----
        if (sub == 0) {
            const char* nxt = (const char*)(logits +
                ((size_t)b * S_DIM + base + TILE_S) * V_DIM);
            // 65 rows x 1 KB ~ 66.5 KB; 128B granularity, speculative (safe OOB)
            __builtin_prefetch(nxt + (tid * 2 + 0) * 128, 0, 0);
            __builtin_prefetch(nxt + (tid * 2 + 1) * 128, 0, 0);
            if (tid < 8) __builtin_prefetch(nxt + 65536 + tid * 128, 0, 0);
        }

        // ---- Phase 2: left = dA_tile @ M via v_wmma_f32_16x16x32_bf16 ----
        // Double-buffered fragments; sched_barrier(0) pins the emitted order:
        //   [18 ds_load_b128 of chunk kc+1] [fence] [8 v_wmma of chunk kc]
        // so waits are partial and LDS latency hides under a chunk of XDL work.
        const __bf16* aBase = dA + arow * MT_STRIDE;
        const __bf16* bBase = Mt + (128 * nh + l16) * MT_STRIDE;

        v8f acc[8] = {};
        ABFrag aF[2];
        ABFrag bF[2][8];

        aF[0].h[0] = *(const v8bf*)(aBase + 16 * half);
        aF[0].h[1] = *(const v8bf*)(aBase + 16 * half + 8);
#pragma unroll
        for (int t = 0; t < 8; ++t) {
            const __bf16* bp = bBase + t * 16 * MT_STRIDE + 16 * half;
            bF[0][t].h[0] = *(const v8bf*)(bp);
            bF[0][t].h[1] = *(const v8bf*)(bp + 8);
        }

#pragma unroll
        for (int kc = 0; kc < 8; ++kc) {
            const int cur = kc & 1;
            const int nxt2 = cur ^ 1;
            if (kc < 7) {
                const int kk = (kc + 1) * 32;
                aF[nxt2].h[0] = *(const v8bf*)(aBase + kk + 16 * half);
                aF[nxt2].h[1] = *(const v8bf*)(aBase + kk + 16 * half + 8);
#pragma unroll
                for (int t = 0; t < 8; ++t) {
                    const __bf16* bp = bBase + t * 16 * MT_STRIDE + kk + 16 * half;
                    bF[nxt2][t].h[0] = *(const v8bf*)(bp);
                    bF[nxt2][t].h[1] = *(const v8bf*)(bp + 8);
                }
            }
            SCHED_FENCE();   // kc+1 loads stay ahead of kc WMMAs
#pragma unroll
            for (int t = 0; t < 8; ++t) {
                acc[t] = __builtin_amdgcn_wmma_f32_16x16x32_bf16(
                    false, aF[cur].v, false, bF[cur][t].v, (short)0, acc[t],
                    false, false);
            }
            SCHED_FENCE();   // kc WMMAs stay ahead of kc+2 loads
        }

        // ---- Phase 3: out[i] = sum_n left[i,n] * d[i+1,n] ----
#pragma unroll
        for (int j = 0; j < 8; ++j) {
            const int r = rowbase + j + 8 * half;
            float p = 0.f;
#pragma unroll
            for (int t = 0; t < 8; ++t) {
                const int n = 128 * nh + 16 * t + l16;
                p += acc[t][j] * (float)dA[(r + 1) * MT_STRIDE + n];
            }
            p += __shfl_xor(p, 1, 32);
            p += __shfl_xor(p, 2, 32);
            p += __shfl_xor(p, 4, 32);
            p += __shfl_xor(p, 8, 32);
            if (l16 == 0) part[nh * TILE_S + r] = p;
        }

        __syncthreads();

        if (tid < TILE_S) {
            const int i = base + tid;
            if (i < S_DIM - 1) {
                out[(size_t)b * (S_DIM - 1) + i] = part[tid] + part[TILE_S + tid];
            }
        }
        // Next sub's dA writes are ordered by the fill-phase __syncthreads.
    }
}

extern "C" void kernel_launch(void* const* d_in, const int* in_sizes, int n_in,
                              void* d_out, int out_size, void* d_ws, size_t ws_size,
                              hipStream_t stream) {
    const float* logits = (const float*)d_in[0];   // [B, S, V] f32
    const float* M      = (const float*)d_in[1];   // [V, V] f32
    float* out          = (float*)d_out;           // [B, S-1] f32

    dim3 grid(S_DIM / (TILE_S * TILES_PER_WG), B_DIM);   // 16 x 64 workgroups

    if (ws_size >= (size_t)MT_BYTES) {
        __bf16* MtG = (__bf16*)d_ws;
        prep_mt_kernel<<<dim3(V_DIM / 4), dim3(256), 0, stream>>>(M, MtG);
        pk_bilinear_wmma_kernel<true><<<grid, dim3(256), SMEM_BYTES, stream>>>(
            logits, M, (const void*)MtG, out);
    } else {
        pk_bilinear_wmma_kernel<false><<<grid, dim3(256), SMEM_BYTES, stream>>>(
            logits, M, nullptr, out);
    }
}
